// SubGraph_49331994362187
// MI455X (gfx1250) — compile-verified, matrix operand order checked by reading, not run on hardware
//
#include <hip/hip_runtime.h>
#include <hip/hip_bf16.h>

// ---------------------------------------------------------------------------
// Problem constants (match reference)
// ---------------------------------------------------------------------------
#define BB 8
#define LL 2048
#define HH 256
#define NHEAD 2
#define HD 128
#define DEPTH 3
#define MTOT (BB * LL)          // 16384 rows of activations

typedef __bf16 bf16_t;
typedef __attribute__((ext_vector_type(8)))  bf16_t v8bf;
typedef __attribute__((ext_vector_type(16))) bf16_t v16bf;
typedef __attribute__((ext_vector_type(8)))  float  v8f;

static __device__ __forceinline__ v16bf mk16(v8bf lo, v8bf hi) {
    v16bf r;
#pragma unroll
    for (int i = 0; i < 8; ++i) { r[i] = lo[i]; r[i + 8] = hi[i]; }
    return r;
}

static __device__ __forceinline__ v8f wmma_bf16(v16bf a, v16bf b, v8f c) {
    // D = A(16x32) * B(32x16) + C, f32 accumulate
    return __builtin_amdgcn_wmma_f32_16x16x32_bf16(
        false, a, false, b, (short)0, c, false, false);
}

static __device__ __forceinline__ v8f vzero8() {
    v8f z = {0.f, 0.f, 0.f, 0.f, 0.f, 0.f, 0.f, 0.f};
    return z;
}

// ---------------------------------------------------------------------------
// Async global -> LDS copy via inline asm (gfx1250 GLOBAL_LOAD_ASYNC_TO_LDS_B128,
// tracked by ASYNCcnt; ISA 15.18.3 op 98). Inline asm sidesteps the toolchain-
// specific builtin signature. LDS operand = wave-relative byte offset (low 32
// bits of the generic pointer). Fallback: synchronous VGPR round-trip.
// ---------------------------------------------------------------------------
#if defined(__gfx1250__)
#define HAS_ASYNC_LDS 1
#else
#define HAS_ASYNC_LDS 0
#endif

static __device__ __forceinline__ void cp16_g2lds(bf16_t* lds_dst, const bf16_t* gsrc) {
#if HAS_ASYNC_LDS
    unsigned lds_off = (unsigned)(uintptr_t)lds_dst;  // LDS byte offset
    asm volatile("global_load_async_to_lds_b128 %0, %1, off"
                 :
                 : "v"(lds_off), "v"(gsrc)
                 : "memory");
#else
    *(v8bf*)lds_dst = *(const v8bf*)gsrc;
#endif
}

static __device__ __forceinline__ void cp_async_wait() {
#if HAS_ASYNC_LDS
    asm volatile("s_wait_asynccnt 0x0" ::: "memory");
#endif
}

// ---------------------------------------------------------------------------
// Weight prep: f32 [K][N] -> bf16 transposed [N][K] (N-major, K contiguous)
// 11 matrices: 0:w0 1:w0a 2-4:wq 5-7:wk 8-10:wv
// ---------------------------------------------------------------------------
__global__ void prep_weights_kernel(const float* __restrict__ w0,
                                    const float* __restrict__ w0a,
                                    const float* __restrict__ wq,
                                    const float* __restrict__ wk,
                                    const float* __restrict__ wv,
                                    bf16_t* __restrict__ wt) {
    int mat = blockIdx.y;
    int n = blockIdx.x;
    int k = threadIdx.x;
    const float* src;
    if (mat == 0)      src = w0;
    else if (mat == 1) src = w0a;
    else if (mat < 5)  src = wq + (mat - 2) * HH * HH;
    else if (mat < 8)  src = wk + (mat - 5) * HH * HH;
    else               src = wv + (mat - 8) * HH * HH;
    wt[(size_t)mat * HH * HH + n * HH + k] = (bf16_t)src[k * HH + n];
}

__global__ void f2bf_kernel(const float* __restrict__ x, bf16_t* __restrict__ o, int n) {
    int i = blockIdx.x * 256 + threadIdx.x;
    if (i < n) o[i] = (bf16_t)x[i];
}

// ---------------------------------------------------------------------------
// GEMM: Y[M=16384][N=256] = X_bf16 @ Wt^T + bias, via v_wmma_f32_16x16x32_bf16
// grid (M/64, N/128), block 256 (8 waves: 4 M-subtiles x 2 N-subtiles)
// mode 0: write f32 Y[m][n]
// mode 1: write bf16 Q/K split per head: [b][head][l][hd]
// mode 2: write bf16 V transposed:       [b][head][hd][l]
// ---------------------------------------------------------------------------
__global__ void gemm_bf16_kernel(const bf16_t* __restrict__ X,
                                 const bf16_t* __restrict__ Wt,
                                 const float* __restrict__ bias,
                                 float* __restrict__ outF,
                                 bf16_t* __restrict__ outB,
                                 int mode) {
    const int tid = threadIdx.x;
    const int lane = tid & 31, wave = tid >> 5;
    const int wm = wave & 3, wn = wave >> 2;
    const int ncol = lane & 15, khalf = lane >> 4;
    const int m0 = blockIdx.x * 64 + wm * 16;
    const int n0 = blockIdx.y * 128 + wn * 64;

    v8f acc[4];
#pragma unroll
    for (int i = 0; i < 4; ++i) acc[i] = vzero8();

    const size_t arow = (size_t)(m0 + ncol) * HH;   // A-operand row for this lane
#pragma unroll
    for (int ks = 0; ks < 8; ++ks) {
        const int off = ks * 32 + khalf * 8;
        v16bf a = mk16(*(const v8bf*)&X[arow + off],
                       *(const v8bf*)&X[arow + off + 16]);
#pragma unroll
        for (int nt = 0; nt < 4; ++nt) {
            const size_t brow = (size_t)(n0 + nt * 16 + ncol) * HH;
            v16bf b = mk16(*(const v8bf*)&Wt[brow + off],
                           *(const v8bf*)&Wt[brow + off + 16]);
            acc[nt] = wmma_bf16(a, b, acc[nt]);
        }
    }
    // Epilogue: lane holds col=ncol, rows g + 8*khalf
#pragma unroll
    for (int nt = 0; nt < 4; ++nt) {
        const int n = n0 + nt * 16 + ncol;
        const float bn = bias[n];
#pragma unroll
        for (int g = 0; g < 8; ++g) {
            const int m = m0 + g + 8 * khalf;
            const float v = acc[nt][g] + bn;
            if (mode == 0) {
                outF[(size_t)m * HH + n] = v;
            } else {
                const int b = m >> 11, lpos = m & (LL - 1);
                const int head = n >> 7, hd = n & (HD - 1);
                if (mode == 1)
                    outB[(((size_t)(b * NHEAD + head)) * LL + lpos) * HD + hd] = (bf16_t)v;
                else
                    outB[(((size_t)(b * NHEAD + head)) * HD + hd) * LL + lpos] = (bf16_t)v;
            }
        }
    }
}

// ---------------------------------------------------------------------------
// LayerNorm kernel. wave per row (8 rows / 256-thread block).
// mode 0:  out = relu(LN(in))                        (MLP)
// mode 1:  out = LN(relu(in) + res)                  (post-attention)
// Writes f32 + bf16 copies of the activation.
// ---------------------------------------------------------------------------
__global__ void ln_kernel(const float* __restrict__ in,
                          const float* __restrict__ res,
                          const float* __restrict__ gam,
                          const float* __restrict__ bet,
                          float* __restrict__ outF,
                          bf16_t* __restrict__ outB,
                          int mode) {
    const int lane = threadIdx.x & 31, wave = threadIdx.x >> 5;
    const int row = blockIdx.x * 8 + wave;
    const size_t base = (size_t)row * HH;
    float x[8];
#pragma unroll
    for (int i = 0; i < 8; ++i) {
        const int c = lane * 8 + i;
        float v = in[base + c];
        if (mode == 1) v = fmaxf(v, 0.f) + res[base + c];
        x[i] = v;
    }
    float s = 0.f;
#pragma unroll
    for (int i = 0; i < 8; ++i) s += x[i];
#pragma unroll
    for (int m = 1; m < 32; m <<= 1) s += __shfl_xor(s, m, 32);
    const float u = s * (1.f / HH);
    float q = 0.f;
#pragma unroll
    for (int i = 0; i < 8; ++i) { const float d = x[i] - u; q += d * d; }
#pragma unroll
    for (int m = 1; m < 32; m <<= 1) q += __shfl_xor(q, m, 32);
    const float inv = rsqrtf(q * (1.f / HH) + 1e-5f);
#pragma unroll
    for (int i = 0; i < 8; ++i) {
        const int c = lane * 8 + i;
        float o = (x[i] - u) * inv * gam[c] + bet[c];
        if (mode == 0) o = fmaxf(o, 0.f);
        outF[base + c] = o;
        outB[base + c] = (bf16_t)o;
    }
}

// ---------------------------------------------------------------------------
// Flash attention. block = 128 threads (4 waves), one (b, head, 64-row q-tile).
// Q kept in registers; K / V^T tiles async-staged into LDS; P via LDS for
// C-layout -> A-layout reshape. Writes ctx f32 [B][L][H].
// ---------------------------------------------------------------------------
#define KSTR 136   // 128 + 8 bf16 pad (conflict avoidance)
#define VSTR 72    // 64 + 8
#define PSTR 72

__global__ void attn_kernel(const bf16_t* __restrict__ q,
                            const bf16_t* __restrict__ kmat,
                            const bf16_t* __restrict__ vT,
                            const long long* __restrict__ lens,
                            float* __restrict__ ctx) {
    __shared__ __align__(16) bf16_t sK[64 * KSTR];
    __shared__ __align__(16) bf16_t sV[128 * VSTR];
    __shared__ __align__(16) bf16_t sP[64 * PSTR];

    const int tid = threadIdx.x, lane = tid & 31, w = tid >> 5;
    const int bid = blockIdx.x;
    const int qt = bid & 31;              // 32 q-tiles of 64 rows
    const int head = (bid >> 5) & (NHEAD - 1);
    const int b = bid >> 6;
    const int len = (int)lens[b];
    const int ncol = lane & 15, khalf = lane >> 4;
    const int qrow0 = qt * 64 + w * 16;

    const size_t hbase = (size_t)(b * NHEAD + head) * LL * HD;  // [L][HD]
    const size_t vbase = (size_t)(b * NHEAD + head) * HD * LL;  // [HD][L]

    // Preload this wave's 16 Q rows as 4 A operands (HD = 4 * 32)
    v16bf qa[4];
    {
        const size_t rb = hbase + (size_t)(qrow0 + ncol) * HD;
#pragma unroll
        for (int ks = 0; ks < 4; ++ks) {
            const int off = ks * 32 + khalf * 8;
            qa[ks] = mk16(*(const v8bf*)&q[rb + off],
                          *(const v8bf*)&q[rb + off + 16]);
        }
    }

    v8f oacc[8];
#pragma unroll
    for (int i = 0; i < 8; ++i) oacc[i] = vzero8();
    float rmax[8], rsum[8];
#pragma unroll
    for (int g = 0; g < 8; ++g) { rmax[g] = -3.0e38f; rsum[g] = 0.f; }

    const float SCALE = 0.08838834764831845f;  // 1/sqrt(128)

    for (int kb = 0; kb < LL / 64; ++kb) {
        __syncthreads();  // previous iteration's LDS readers done
        // ---- async K tile: 64 keys x 128 hd (8 x b128 per thread) ----
        for (int i = tid; i < 1024; i += 128) {
            const int r = i >> 4, c = i & 15;
            cp16_g2lds(&sK[r * KSTR + c * 8],
                       &kmat[hbase + (size_t)(kb * 64 + r) * HD + c * 8]);
        }
        // ---- async V^T tile: 128 hd x 64 keys ----
        for (int i = tid; i < 1024; i += 128) {
            const int r = i >> 3, c = i & 7;
            cp16_g2lds(&sV[r * VSTR + c * 8],
                       &vT[vbase + (size_t)r * LL + kb * 64 + c * 8]);
        }
        cp_async_wait();   // s_wait_asynccnt 0 (own transfers complete)
        __syncthreads();   // all threads' transfers complete

        // ---- S = Q * K^T (64 keys) : 16 WMMAs / wave ----
        v8f sacc[4];
#pragma unroll
        for (int nt = 0; nt < 4; ++nt) sacc[nt] = vzero8();
#pragma unroll
        for (int ks = 0; ks < 4; ++ks) {
            const int off = ks * 32 + khalf * 8;
#pragma unroll
            for (int nt = 0; nt < 4; ++nt) {
                const int r = nt * 16 + ncol;
                v16bf bm = mk16(*(const v8bf*)&sK[r * KSTR + off],
                                *(const v8bf*)&sK[r * KSTR + off + 16]);
                sacc[nt] = wmma_bf16(qa[ks], bm, sacc[nt]);
            }
        }

        // ---- online softmax ----
        float cur[4][8];
#pragma unroll
        for (int nt = 0; nt < 4; ++nt) {
            const int key = kb * 64 + nt * 16 + ncol;
#pragma unroll
            for (int g = 0; g < 8; ++g) {
                float sv = sacc[nt][g] * SCALE;
                const int qr = qrow0 + g + 8 * khalf;
                if (qr < len && key >= len) sv -= 10000.f;
                cur[nt][g] = sv;
            }
        }
        float mnew[8], corr[8], psum[8];
#pragma unroll
        for (int g = 0; g < 8; ++g) {
            float mg = fmaxf(fmaxf(cur[0][g], cur[1][g]), fmaxf(cur[2][g], cur[3][g]));
#pragma unroll
            for (int m = 1; m < 16; m <<= 1) mg = fmaxf(mg, __shfl_xor(mg, m, 32));
            mnew[g] = fmaxf(rmax[g], mg);
            corr[g] = __expf(rmax[g] - mnew[g]);
            rmax[g] = mnew[g];
            psum[g] = 0.f;
        }
#pragma unroll
        for (int h = 0; h < 8; ++h)
#pragma unroll
            for (int g = 0; g < 8; ++g) oacc[h][g] *= corr[g];
#pragma unroll
        for (int nt = 0; nt < 4; ++nt) {
#pragma unroll
            for (int g = 0; g < 8; ++g) {
                const float p = __expf(cur[nt][g] - mnew[g]);
                psum[g] += p;
                sP[(w * 16 + g + 8 * khalf) * PSTR + nt * 16 + ncol] = (bf16_t)p;
            }
        }
#pragma unroll
        for (int g = 0; g < 8; ++g) {
#pragma unroll
            for (int m = 1; m < 16; m <<= 1) psum[g] += __shfl_xor(psum[g], m, 32);
            rsum[g] = rsum[g] * corr[g] + psum[g];
        }
        __syncthreads();  // P visible to all lanes

        // ---- O += P * V : 16 WMMAs / wave ----
#pragma unroll
        for (int kp = 0; kp < 2; ++kp) {
            const int off = kp * 32 + khalf * 8;
            const int pr = (w * 16 + ncol) * PSTR;
            v16bf pa = mk16(*(const v8bf*)&sP[pr + off],
                            *(const v8bf*)&sP[pr + off + 16]);
#pragma unroll
            for (int ht = 0; ht < 8; ++ht) {
                const int r = ht * 16 + ncol;
                v16bf bm = mk16(*(const v8bf*)&sV[r * VSTR + off],
                                *(const v8bf*)&sV[r * VSTR + off + 16]);
                oacc[ht] = wmma_bf16(pa, bm, oacc[ht]);
            }
        }
    }

    // ---- epilogue: normalize + store ctx [B][L][H] ----
    float inv[8];
#pragma unroll
    for (int g = 0; g < 8; ++g) inv[g] = 1.f / rsum[g];
#pragma unroll
    for (int ht = 0; ht < 8; ++ht) {
#pragma unroll
        for (int g = 0; g < 8; ++g) {
            const int qr = qrow0 + g + 8 * khalf;
            ctx[(size_t)(b * LL + qr) * HH + head * HD + ht * 16 + ncol] =
                oacc[ht][g] * inv[g];
        }
    }
}

// ---------------------------------------------------------------------------
// pooled = max over padded L (matches reference), out[0 .. B*H)
// ---------------------------------------------------------------------------
__global__ void pool_kernel(const float* __restrict__ h, float* __restrict__ out) {
    const int b = blockIdx.x, t = threadIdx.x;
    float m = -3.4e38f;
    for (int l = 0; l < LL; ++l) {
        if (l + 8 < LL)
            __builtin_prefetch(&h[(size_t)(b * LL + l + 8) * HH + t], 0, 1);
        m = fmaxf(m, h[(size_t)(b * LL + l) * HH + t]);
    }
    out[b * HH + t] = m;
}

// ---------------------------------------------------------------------------
// cat = gather valid rows, out[B*H ...]. Prefix offsets computed inline (B=8).
// ---------------------------------------------------------------------------
__global__ void gather_kernel(const float* __restrict__ h,
                              const long long* __restrict__ lens,
                              float* __restrict__ out) {
    const int i = blockIdx.x;
    const int b = i >> 11, l = i & (LL - 1);
    const long long lb = lens[b];
    if ((long long)l >= lb) return;
    long long off = 0;
    for (int j = 0; j < b; ++j) off += lens[j];
    const size_t dst = (size_t)BB * HH + (size_t)(off + l) * HH;
    out[dst + threadIdx.x] = h[(size_t)i * HH + threadIdx.x];
}

// ---------------------------------------------------------------------------
// Host launcher
// ---------------------------------------------------------------------------
extern "C" void kernel_launch(void* const* d_in, const int* in_sizes, int n_in,
                              void* d_out, int out_size, void* d_ws, size_t ws_size,
                              hipStream_t stream) {
    const float*     x     = (const float*)d_in[0];
    const long long* lens  = (const long long*)d_in[1];   // int64 lengths
    const float*     w0    = (const float*)d_in[2];
    const float*     b0    = (const float*)d_in[3];
    const float*     ln0g  = (const float*)d_in[4];
    const float*     ln0b  = (const float*)d_in[5];
    const float*     w0a   = (const float*)d_in[6];
    const float*     b0a   = (const float*)d_in[7];
    const float*     ln0ag = (const float*)d_in[8];
    const float*     ln0ab = (const float*)d_in[9];
    const float*     wq    = (const float*)d_in[10];
    const float*     bq    = (const float*)d_in[11];
    const float*     wk    = (const float*)d_in[12];
    const float*     bk    = (const float*)d_in[13];
    const float*     wv    = (const float*)d_in[14];
    const float*     bv    = (const float*)d_in[15];
    const float*     lng   = (const float*)d_in[16];
    const float*     lnb   = (const float*)d_in[17];
    float* out = (float*)d_out;

    // Workspace layout
    const size_t ACT = (size_t)MTOT * HH;            // 4,194,304 elements
    float*  y_f32 = (float*)d_ws;                    // GEMM out (pre-LN)
    float*  h_f32 = y_f32 + ACT;
    float*  ctxf  = h_f32 + ACT;
    bf16_t* hb    = (bf16_t*)(ctxf + ACT);
    bf16_t* qb    = hb + ACT;
    bf16_t* kbuf  = qb + (size_t)BB * NHEAD * LL * HD;
    bf16_t* vtb   = kbuf + (size_t)BB * NHEAD * LL * HD;
    bf16_t* wt    = vtb + (size_t)BB * NHEAD * LL * HD;  // 11 * 256 * 256 bf16

    const dim3 gemm_grid(MTOT / 64, HH / 128);

    // 0) weight prep + input conversion
    prep_weights_kernel<<<dim3(HH, 11), HH, 0, stream>>>(w0, w0a, wq, wk, wv, wt);
    f2bf_kernel<<<(int)(ACT / 256), 256, 0, stream>>>(x, hb, (int)ACT);

    // 1) MLP 0: relu(LN(x @ w0 + b0))
    gemm_bf16_kernel<<<gemm_grid, 256, 0, stream>>>(hb, wt + 0 * HH * HH, b0, y_f32, nullptr, 0);
    ln_kernel<<<MTOT / 8, 256, 0, stream>>>(y_f32, nullptr, ln0g, ln0b, h_f32, hb, 0);

    // 2) MLP 0a
    gemm_bf16_kernel<<<gemm_grid, 256, 0, stream>>>(hb, wt + 1 * HH * HH, b0a, y_f32, nullptr, 0);
    ln_kernel<<<MTOT / 8, 256, 0, stream>>>(y_f32, nullptr, ln0ag, ln0ab, h_f32, hb, 0);

    // 3) attention depth loop
    for (int d = 0; d < DEPTH; ++d) {
        gemm_bf16_kernel<<<gemm_grid, 256, 0, stream>>>(hb, wt + (2 + d) * HH * HH, bq + d * HH, nullptr, qb, 1);
        gemm_bf16_kernel<<<gemm_grid, 256, 0, stream>>>(hb, wt + (5 + d) * HH * HH, bk + d * HH, nullptr, kbuf, 1);
        gemm_bf16_kernel<<<gemm_grid, 256, 0, stream>>>(hb, wt + (8 + d) * HH * HH, bv + d * HH, nullptr, vtb, 2);
        attn_kernel<<<BB * NHEAD * (LL / 64), 128, 0, stream>>>(qb, kbuf, vtb, lens, ctxf);
        ln_kernel<<<MTOT / 8, 256, 0, stream>>>(ctxf, h_f32, lng + d * HH, lnb + d * HH, h_f32, hb, 1);
    }

    // 4) outputs: pooled max + variable-length gather
    pool_kernel<<<BB, HH, 0, stream>>>(h_f32, out);
    gather_kernel<<<MTOT, HH, 0, stream>>>(h_f32, lens, out);
}
